// KMeans_60911226192342
// MI455X (gfx1250) — matrix operand order repaired and checked by reading, not compiled, not run on hardware
//
#include <hip/hip_runtime.h>
#include <hip/hip_bf16.h>

typedef __attribute__((ext_vector_type(16))) _Float16 v16h;
typedef __attribute__((ext_vector_type(8)))  _Float16 v8h;
typedef __attribute__((ext_vector_type(8)))  float    v8f;

#define NROWS 131072
#define HID   128
#define KTOT  1152      // 128 base + 128*8 spline features
#define NPROTO 256
#define XS_STRIDE 130   // f16 pad
#define FS_STRIDE 40    // 32 + pad, keeps 16B alignment (80B rows)
#define EMB_STRIDE 136  // 128 + pad, 272B rows (16B aligned)

// ---------------------------------------------------------------- helpers
__device__ __forceinline__ float silu_f(float x) {
    return x / (1.0f + __expf(-x));
}

// ------------------------------------------------- weight prepack (f16)
// Wt[o][k], k<128 -> base_w[o][k]; k>=128 -> spline_w[o][i][c]*scaler[o][i]
__global__ __launch_bounds__(256) void prep_weights(
    const float* bw1, const float* sw1, const float* sc1,
    const float* bw2, const float* sw2, const float* sc2,
    _Float16* Wt1, _Float16* Wt2)
{
    int idx = blockIdx.x * 256 + threadIdx.x;     // 0 .. 2*147456-1
    int layer = (idx >= HID * KTOT) ? 1 : 0;
    int r = idx - layer * HID * KTOT;
    int o = r / KTOT;
    int k = r - o * KTOT;
    const float* bw = layer ? bw2 : bw1;
    const float* sw = layer ? sw2 : sw1;
    const float* sc = layer ? sc2 : sc1;
    float v;
    if (k < HID) {
        v = bw[o * HID + k];
    } else {
        int i = (k - HID) >> 3, c = (k - HID) & 7;
        v = sw[(o * HID + i) * 8 + c] * sc[o * HID + i];
    }
    (layer ? Wt2 : Wt1)[o * KTOT + k] = (_Float16)v;
}

// ------------------------------------------------- prototype normalize
__global__ __launch_bounds__(128) void prep_protos(const float* protos, _Float16* pn) {
    int p = blockIdx.x, t = threadIdx.x;          // 128 threads
    float v = protos[p * HID + t];
    float s = v * v;
#pragma unroll
    for (int off = 16; off > 0; off >>= 1) s += __shfl_xor(s, off, 32);
    __shared__ float part[4];
    if ((t & 31) == 0) part[t >> 5] = s;
    __syncthreads();
    float norm = sqrtf(part[0] + part[1] + part[2] + part[3]);
    norm = fmaxf(norm, 1e-8f);
    pn[p * HID + t] = (_Float16)(v / norm);
}

// ------------------------------------------------- fused KANLinear layer
// One block = 128 rows. 8 waves, each owns a 16-row band x 128 output cols.
__global__ __launch_bounds__(256) void kan_layer(
    const float* in, const _Float16* Wt, float* out)
{
    __shared__ _Float16 xs[HID][XS_STRIDE];   // input rows, f16
    __shared__ _Float16 Fs[HID][FS_STRIDE];   // feature tile (128 rows x 32 k)
    __shared__ _Float16 Bs[HID][FS_STRIDE];   // weight panel  (128 out  x 32 k)

    const int tid = threadIdx.x;
    const long rowbase = (long)blockIdx.x * HID;

    for (int e = tid; e < HID * HID; e += 256) {
        int r = e >> 7, c = e & 127;
        xs[r][c] = (_Float16)in[(rowbase + r) * HID + c];
    }
    __syncthreads();

    const int wave = tid >> 5, lane = tid & 31;
    const int arow = wave * 16 + (lane & 15);
    const int kA = (lane >> 4) * 8;     // A frag: K {0..7,16..23} / {8..15,24..31}
    const int kB = (lane >> 4) * 8;     // B frag: K {0..7,16..23} / {8..15,24..31}
    const int bcol = lane & 15;

    v8f zero = {};
    v8f acc[8];
#pragma unroll
    for (int t = 0; t < 8; ++t) acc[t] = zero;

    const int ostage = tid >> 1;                  // weight-panel staging coords
    const int kstage = (tid & 1) * 16;

    for (int k0 = 0; k0 < KTOT; k0 += 32) {
        __syncthreads();                          // protect Fs/Bs reuse
        // stage 32x128 weight panel (transposed: Bs[o][kk])
        {
            const _Float16* src = Wt + ostage * KTOT + k0 + kstage;
            *(v8h*)&Bs[ostage][kstage]     = *(const v8h*)(src);
            *(v8h*)&Bs[ostage][kstage + 8] = *(const v8h*)(src + 8);
            if (k0 + 32 < KTOT)                   // pull next panel toward L2/L0
                __builtin_prefetch(src + 32, 0, 0);
        }
        // compute feature tile
        if (k0 < HID) {                            // silu(x) region
            int r = tid >> 1, kk0 = (tid & 1) * 16;
#pragma unroll
            for (int kk = 0; kk < 16; ++kk) {
                float x = (float)xs[r][k0 + kk0 + kk];
                Fs[r][kk0 + kk] = (_Float16)silu_f(x);
            }
        } else {
            // b-spline region: 4 input features x 8 coefs per K-step.
            // Uniform grid -> closed-form cubic B-spline: exactly 4 nonzero
            // taps at basis indices m-3..m, m = interval of x.
            int i0 = (k0 - HID) >> 3;
#pragma unroll
            for (int task = tid; task < 512; task += 256) {
                int r = task & 127, fi = task >> 7;
                float x = (float)xs[r][i0 + fi];
                float s = (x + 2.2f) * 2.5f;       // (x - t_0) / h
                float fm = floorf(s);
                float u = s - fm;
                int m = (int)fm;
                bool inr = (s >= 0.0f) && (s < 11.0f);
                float u2 = u * u, u3 = u2 * u;
                const float k6 = 1.0f / 6.0f;
                float om = 1.0f - u;
                float w_m  = u3 * k6;                                   // j = m
                float w_m1 = (-3.0f*u3 + 3.0f*u2 + 3.0f*u + 1.0f) * k6; // j = m-1
                float w_m2 = ( 3.0f*u3 - 6.0f*u2 + 4.0f) * k6;          // j = m-2
                float w_m3 = om * om * om * k6;                         // j = m-3
                v8h pk;
#pragma unroll
                for (int c = 0; c < 8; ++c) {
                    int d = m - c;
                    float v = (d == 0) ? w_m
                            : (d == 1) ? w_m1
                            : (d == 2) ? w_m2
                            : (d == 3) ? w_m3 : 0.0f;
                    pk[c] = (_Float16)(inr ? v : 0.0f);
                }
                *(v8h*)&Fs[r][fi * 8] = pk;
            }
        }
        __syncthreads();

        // A fragment for this wave's 16-row band
        v8h alo = *(const v8h*)&Fs[arow][kA];
        v8h ahi = *(const v8h*)&Fs[arow][kA + 16];
        v16h a;
#pragma unroll
        for (int e = 0; e < 8; ++e) { a[e] = alo[e]; a[e + 8] = ahi[e]; }

#pragma unroll
        for (int t = 0; t < 8; ++t) {
            int o = t * 16 + bcol;
            v8h blo = *(const v8h*)&Bs[o][kB];
            v8h bhi = *(const v8h*)&Bs[o][kB + 16];
            v16h b;
#pragma unroll
            for (int e = 0; e < 8; ++e) { b[e] = blo[e]; b[e + 8] = bhi[e]; }
            acc[t] = __builtin_amdgcn_wmma_f32_16x16x32_f16(
                false, a, false, b, (short)0, acc[t], false, false);
        }
    }

    // epilogue: C/D layout -> lane l, vgpr r => row = r + (l>=16 ? 8:0), col = l&15
    const long orow0 = rowbase + wave * 16 + (lane >> 4) * 8;
#pragma unroll
    for (int t = 0; t < 8; ++t) {
        int col = t * 16 + (lane & 15);
#pragma unroll
        for (int r = 0; r < 8; ++r) {
            out[(orow0 + r) * HID + col] = acc[t][r];
        }
    }
}

// ------------------------------------------------- sim + argmax
// argmax of cos-sim == argmax of emb @ pn^T (pn pre-normalized; row scale > 0)
__global__ __launch_bounds__(256) void sim_argmax(
    const float* emb, const _Float16* pn, float* assign)
{
    __shared__ _Float16 xs[HID][EMB_STRIDE];
    const int tid = threadIdx.x;
    const long rowbase = (long)blockIdx.x * HID;

    for (int e = tid; e < HID * HID; e += 256) {
        int r = e >> 7, c = e & 127;
        xs[r][c] = (_Float16)emb[(rowbase + r) * HID + c];
    }
    __syncthreads();

    const int wave = tid >> 5, lane = tid & 31;
    const int arow = wave * 16 + (lane & 15);
    const int kA = (lane >> 4) * 8;
    const int kB = (lane >> 4) * 8;
    const int pcol = lane & 15;

    // preload all 4 A fragments (K = 128)
    v16h a[4];
#pragma unroll
    for (int ks = 0; ks < 4; ++ks) {
        v8h lo = *(const v8h*)&xs[arow][ks * 32 + kA];
        v8h hi = *(const v8h*)&xs[arow][ks * 32 + kA + 16];
#pragma unroll
        for (int e = 0; e < 8; ++e) { a[ks][e] = lo[e]; a[ks][e + 8] = hi[e]; }
    }

    float best[8];
    int   bidx[8];
#pragma unroll
    for (int r = 0; r < 8; ++r) { best[r] = -3.4e38f; bidx[r] = 0; }

    v8f zero = {};
    for (int t = 0; t < 16; ++t) {                 // 16 tiles of 16 prototypes
        v8f acc = zero;
        const _Float16* prow = pn + (t * 16 + pcol) * HID;
#pragma unroll
        for (int ks = 0; ks < 4; ++ks) {
            v8h blo = *(const v8h*)(prow + ks * 32 + kB);
            v8h bhi = *(const v8h*)(prow + ks * 32 + kB + 16);
            v16h b;
#pragma unroll
            for (int e = 0; e < 8; ++e) { b[e] = blo[e]; b[e + 8] = bhi[e]; }
            acc = __builtin_amdgcn_wmma_f32_16x16x32_f16(
                false, a[ks], false, b, (short)0, acc, false, false);
        }
        int idx = t * 16 + pcol;
#pragma unroll
        for (int r = 0; r < 8; ++r) {
            float v = acc[r];
            if (v > best[r] || (v == best[r] && idx < bidx[r])) {
                best[r] = v; bidx[r] = idx;
            }
        }
    }

    // reduce across the 16 lanes (columns) of each half-wave; ties -> lowest idx
#pragma unroll
    for (int off = 8; off >= 1; off >>= 1) {
#pragma unroll
        for (int r = 0; r < 8; ++r) {
            float ov = __shfl_xor(best[r], off, 32);
            int   oi = __shfl_xor(bidx[r], off, 32);
            if (ov > best[r] || (ov == best[r] && oi < bidx[r])) {
                best[r] = ov; bidx[r] = oi;
            }
        }
    }
    if ((lane & 15) == 0) {
        long rb = rowbase + wave * 16 + (lane >> 4) * 8;
#pragma unroll
        for (int r = 0; r < 8; ++r) assign[rb + r] = (float)bidx[r];
    }
}

// ------------------------------------------------- launcher
extern "C" void kernel_launch(void* const* d_in, const int* in_sizes, int n_in,
                              void* d_out, int out_size, void* d_ws, size_t ws_size,
                              hipStream_t stream) {
    const float* x      = (const float*)d_in[0];
    const float* protos = (const float*)d_in[1];
    // d_in[2] = grid: uniform, recomputed analytically in-kernel
    const float* bw1 = (const float*)d_in[3];
    const float* sw1 = (const float*)d_in[4];
    const float* sc1 = (const float*)d_in[5];
    const float* bw2 = (const float*)d_in[6];
    const float* sw2 = (const float*)d_in[7];
    const float* sc2 = (const float*)d_in[8];

    _Float16* Wt1 = (_Float16*)d_ws;              // 128*1152 f16
    _Float16* Wt2 = Wt1 + HID * KTOT;             // 128*1152 f16
    _Float16* pn  = Wt2 + HID * KTOT;             // 256*128  f16

    float* emb    = (float*)d_out;                // N*128
    float* assign = emb + (size_t)NROWS * HID;    // N

    prep_weights<<<(2 * HID * KTOT) / 256, 256, 0, stream>>>(
        bw1, sw1, sc1, bw2, sw2, sc2, Wt1, Wt2);
    prep_protos<<<NPROTO, 128, 0, stream>>>(protos, pn);

    kan_layer<<<NROWS / HID, 256, 0, stream>>>(x,   Wt1, emb);
    kan_layer<<<NROWS / HID, 256, 0, stream>>>(emb, Wt2, emb);   // in-place: block-local rows
    sim_argmax<<<NROWS / HID, 256, 0, stream>>>(emb, pn, assign);
}
